// FeatureFusionBlock_19971597926436
// MI455X (gfx1250) — matrix operand order repaired and checked by reading, not compiled, not run on hardware
//
#include <hip/hip_runtime.h>
#include <hip/hip_bf16.h>
#include <cstdint>

// Problem constants (match reference)
#define BB 16
#define PP 784
#define CC 10
#define MM 4096
#define MG 128
#define DX 1152
#define DR 768
#define WAVES 8     // waves in 256-thread helper kernels
#define GWAVES 8    // waves per block in the WMMA kernel (1 patch-tile each)
#define NTB 8       // bank N-tiles staged in LDS per group (128 rows)

typedef __attribute__((ext_vector_type(16))) __bf16 v16bf;
typedef __attribute__((ext_vector_type(8)))  float  v8f;

// ---------------------------------------------------------------------------
// Row L2 norms: one wave per row, lanes stride columns (coalesced), shuffle-
// reduce. Used for bank norms (yy) and feature norms (xx).
// ---------------------------------------------------------------------------
__global__ __launch_bounds__(256)
void rownorm_kernel(const float* __restrict__ src, float* __restrict__ dst,
                    int nrows, int D) {
    const int wave = threadIdx.x >> 5;
    const int lane = threadIdx.x & 31;
    const int row  = blockIdx.x * WAVES + wave;
    if (row >= nrows) return;
    const float* p = src + (size_t)row * D;
    float s = 0.0f;
    for (int i = lane; i < D; i += 32) { float v = p[i]; s += v * v; }
    #pragma unroll
    for (int m = 16; m >= 1; m >>= 1) s += __shfl_xor(s, m, 32);
    if (lane == 0) dst[row] = s;
}

// ---------------------------------------------------------------------------
// Per-sample prep: nvalid, masked-mean global features g_pt [B,DX], g_rgb [B,DR]
// One block per sample.
// ---------------------------------------------------------------------------
__global__ __launch_bounds__(256)
void meanprep_kernel(const float* __restrict__ fxyz, const float* __restrict__ frgb,
                     const int* __restrict__ mask,
                     float* __restrict__ g_pt, float* __restrict__ g_rgb,
                     float* __restrict__ nvalid) {
    const int b = blockIdx.x, tid = threadIdx.x;
    __shared__ float sm[PP];
    __shared__ float red[256];
    for (int p = tid; p < PP; p += 256) sm[p] = (float)mask[b * PP + p];
    __syncthreads();
    float s = 0.0f;
    for (int p = tid; p < PP; p += 256) s += sm[p];
    red[tid] = s; __syncthreads();
    for (int off = 128; off > 0; off >>= 1) {
        if (tid < off) red[tid] += red[tid + off];
        __syncthreads();
    }
    const float nv = red[0];
    if (tid == 0) nvalid[b] = nv;
    const float inv = 1.0f / nv;
    for (int c = tid; c < DX; c += 256) {
        float acc = 0.0f;
        const float* base = fxyz + (size_t)b * PP * DX + c;
        for (int p = 0; p < PP; ++p) acc += base[(size_t)p * DX] * sm[p];
        g_pt[(size_t)b * DX + c] = acc * inv;
    }
    for (int c = tid; c < DR; c += 256) {
        float acc = 0.0f;
        const float* base = frgb + (size_t)b * PP * DR + c;
        for (int p = 0; p < PP; ++p) acc += base[(size_t)p * DR] * sm[p];
        g_rgb[(size_t)b * DR + c] = acc * inv;
    }
}

// ---------------------------------------------------------------------------
// Main WMMA kernel. Per block: 8 waves, each owning one 16-patch M-tile
// (block covers 128 patches -> only 7 blocks/sample stream the bank).
// NTB=8 bank N-tiles (128 rows x 32 K) are staged in LDS as bf16 once per
// (group, K-chunk) and shared by all 8 waves: 8 WMMAs per A-fragment load.
// Computes minD[p] = sqrt(clamp(xx[p] + min_j (yy[j] - 2 <x_p,y_j>),1e-12,50)).
// bf16 WMMA 16x16x32, fp32 accumulate.
// ---------------------------------------------------------------------------
__global__ __launch_bounds__(256)
void gemm_min_kernel(const float* __restrict__ feat,      // [B,P,D]
                     const float* __restrict__ bank,      // [C,M,D]
                     const float* __restrict__ bankNorms, // [C,M]
                     const float* __restrict__ xx,        // [B,P]
                     const int* __restrict__ labels,      // [B]
                     float* __restrict__ minD,            // [B,P]
                     int D) {
    const int b    = blockIdx.y;
    const int tid  = threadIdx.x;
    const int wave = tid >> 5;
    const int lane = tid & 31;
    const int half = lane >> 4;       // 0: lanes 0-15, 1: lanes 16-31
    const int lrow = lane & 15;
    const int cls  = labels[b];

    const int p0   = (blockIdx.x * GWAVES + wave) * 16;   // this wave's M-tile
    const int pr   = p0 + lrow;
    const int prow = (pr < PP) ? pr : (PP - 1);           // clamp tail (EXEC stays all-1)

    const float* featRow  = feat + ((size_t)b * PP + prow) * D;
    const float* bankBase = bank + (size_t)cls * MM * D;
    const float* normBase = bankNorms + (size_t)cls * MM;

    // B staging: NTB*16 = 128 bank rows x 32 bf16 (16 dwords) + 4 dword pad.
    // Row stride 20 dwords: 16B-aligned for b128 and conflict-free across rows.
    __shared__ __align__(16) uint32_t Bs[NTB * 16][20];

    // cooperative-fill mapping: 128 rows x 2 K-halves = 256 units, 1/thread
    const int ldR = tid >> 1;   // 0..127
    const int ldH = tid & 1;    // 0..1

    float runmin[8];
    #pragma unroll
    for (int r = 0; r < 8; ++r) runmin[r] = 3.0e38f;

    const int ngroups = MM / (16 * NTB);   // 32
    for (int g = 0; g < ngroups; ++g) {
        const int j0 = g * (16 * NTB);
        v8f acc[NTB];
        #pragma unroll
        for (int t = 0; t < NTB; ++t) acc[t] = {};

        for (int kb = 0; kb < D; kb += 32) {
            // --- cooperative fill: 128 rows x 32 floats -> bf16 in LDS
            {
                const float* src = bankBase + (size_t)(j0 + ldR) * D + kb + ldH * 16;
                if (kb + 32 < D)
                    __builtin_prefetch(src + 32, 0, 1);   // global_prefetch_b8
                union { __bf16 h[16]; uint4 q[2]; } pk;
                #pragma unroll
                for (int i = 0; i < 16; ++i) pk.h[i] = (__bf16)src[i];
                uint4* dst = (uint4*)&Bs[ldR][ldH * 8];
                dst[0] = pk.q[0];
                dst[1] = pk.q[1];
            }
            __syncthreads();

            // --- A fragment (16x32 bf16) straight from global: lane half h
            // holds K in [kb+8h, kb+8h+8) (VGPR0-3), [kb+16+8h, +8) (VGPR4-7)
            v16bf a;
            const float* s0 = featRow + kb + half * 8;
            const float* s1 = featRow + kb + 16 + half * 8;
            #pragma unroll
            for (int i = 0; i < 8; ++i) a[i]     = (__bf16)s0[i];
            #pragma unroll
            for (int i = 0; i < 8; ++i) a[8 + i] = (__bf16)s1[i];

            // --- NTB WMMAs; B fragment = 2x ds_load_b128 from staged LDS.
            // B layout: col = bank row (lane%16); half h holds K [kb+8h, +8), [kb+16+8h, +8)
            #pragma unroll
            for (int t = 0; t < NTB; ++t) {
                union { uint4 q[2]; v16bf v; } bf;
                const uint4* ps = (const uint4*)&Bs[t * 16 + lrow][half * 4];
                bf.q[0] = ps[0];
                bf.q[1] = ps[2];
                acc[t] = __builtin_amdgcn_wmma_f32_16x16x32_bf16(
                    false, a, false, bf.v, (short)0, acc[t], false, false);
            }
            __syncthreads();
        }

        // Epilogue: C layout — VGPR r, lane: (row = r + 8*half, col = lane%16)
        #pragma unroll
        for (int t = 0; t < NTB; ++t) {
            const int j  = j0 + t * 16 + lrow;
            const float yy = normBase[j];
            #pragma unroll
            for (int r = 0; r < 8; ++r)
                runmin[r] = fminf(runmin[r], yy - 2.0f * acc[t][r]);
        }
    }

    // Row-wise min across the 16 columns held by each 16-lane half (butterfly)
    #pragma unroll
    for (int r = 0; r < 8; ++r) {
        float v = runmin[r];
        #pragma unroll
        for (int m = 1; m < 16; m <<= 1) v = fminf(v, __shfl_xor(v, m, 32));
        runmin[r] = v;
    }

    if (p0 < PP) {
        const float* xb = xx + (size_t)b * PP;
        float* ob = minD + (size_t)b * PP;
        if (lane == 0) {
            #pragma unroll
            for (int r = 0; r < 8; ++r) {
                const int p = p0 + r;                       // rows 0-7
                const float d2 = xb[p] + runmin[r];
                ob[p] = sqrtf(fminf(fmaxf(d2, 1e-12f), 50.0f));
            }
        }
        if (lane == 16) {
            #pragma unroll
            for (int r = 0; r < 8; ++r) {
                const int p = p0 + 8 + r;                   // rows 8-15
                const float d2 = xb[p] + runmin[r];
                ob[p] = sqrtf(fminf(fmaxf(d2, 1e-12f), 50.0f));
            }
        }
    }
}

// ---------------------------------------------------------------------------
// Global-feature branch: min over MG=128 bank rows of ||g - y||, per sample.
// ---------------------------------------------------------------------------
__global__ __launch_bounds__(256)
void global_min_kernel(const float* __restrict__ g,      // [B,D]
                       const float* __restrict__ bankg,  // [C,MG,D]
                       const int* __restrict__ labels,
                       float* __restrict__ outMin,       // [B]
                       int D) {
    const int b = blockIdx.x;
    const int wave = threadIdx.x >> 5;
    const int lane = threadIdx.x & 31;
    const int cls = labels[b];
    const float* gv = g + (size_t)b * D;
    __shared__ float wmin[WAVES];
    float best = 3.0e38f;
    for (int j = wave; j < MG; j += WAVES) {
        const float* y = bankg + ((size_t)cls * MG + j) * D;
        float s = 0.0f;
        for (int i = lane; i < D; i += 32) { float dd = gv[i] - y[i]; s += dd * dd; }
        #pragma unroll
        for (int m = 16; m >= 1; m >>= 1) s += __shfl_xor(s, m, 32);
        best = fminf(best, s);
    }
    if (lane == 0) wmin[wave] = best;
    __syncthreads();
    if (threadIdx.x == 0) {
        float v = wmin[0];
        #pragma unroll
        for (int w = 1; w < WAVES; ++w) v = fminf(v, wmin[w]);
        outMin[b] = sqrtf(fminf(fmaxf(v, 1e-12f), 50.0f));
    }
}

// ---------------------------------------------------------------------------
// Per-sample term: masked mean of minD_pt, mean of minD_rgb, + global terms.
// ---------------------------------------------------------------------------
__global__ __launch_bounds__(256)
void sample_term_kernel(const float* __restrict__ minD_pt,
                        const float* __restrict__ minD_rgb,
                        const int* __restrict__ mask,
                        const float* __restrict__ nvalid,
                        const float* __restrict__ ming_pt,
                        const float* __restrict__ ming_rgb,
                        float* __restrict__ term) {
    const int b = blockIdx.x, tid = threadIdx.x;
    __shared__ float rp[256], rr[256];
    float sp = 0.0f, sr = 0.0f;
    for (int p = tid; p < PP; p += 256) {
        sp += minD_pt[(size_t)b * PP + p] * (float)mask[b * PP + p];
        sr += minD_rgb[(size_t)b * PP + p];
    }
    rp[tid] = sp; rr[tid] = sr; __syncthreads();
    for (int off = 128; off > 0; off >>= 1) {
        if (tid < off) { rp[tid] += rp[tid + off]; rr[tid] += rr[tid + off]; }
        __syncthreads();
    }
    if (tid == 0) {
        term[b] = rp[0] / nvalid[b] + rr[0] / (float)PP
                + 0.4f * (ming_pt[b] + ming_rgb[b]);
    }
}

__global__ void final_kernel(const float* __restrict__ term, float* __restrict__ out) {
    if (threadIdx.x == 0) {
        float acc = 0.0f;
        for (int b = 0; b < BB; ++b) acc += term[b];
        out[0] = (acc / (float)BB) * 0.9f;
    }
}

// ---------------------------------------------------------------------------
extern "C" void kernel_launch(void* const* d_in, const int* in_sizes, int n_in,
                              void* d_out, int out_size, void* d_ws, size_t ws_size,
                              hipStream_t stream) {
    (void)in_sizes; (void)n_in; (void)out_size; (void)ws_size;
    const float* fxyz = (const float*)d_in[0];   // [B,P,DX]
    const float* frgb = (const float*)d_in[1];   // [B,P,DR]
    const float* bpt  = (const float*)d_in[2];   // [C,M,DX]
    const float* brg  = (const float*)d_in[3];   // [C,M,DR]
    const float* bgp  = (const float*)d_in[4];   // [C,MG,DX]
    const float* bgr  = (const float*)d_in[5];   // [C,MG,DR]
    const int*   lbl  = (const int*)d_in[6];     // [B]
    const int*   msk  = (const int*)d_in[7];     // [B,P]
    float* out = (float*)d_out;

    // Workspace layout (floats)
    float* ws        = (float*)d_ws;
    float* norms_pt  = ws;                       // C*M
    float* norms_rgb = norms_pt + CC * MM;       // C*M
    float* xx_pt     = norms_rgb + CC * MM;      // B*P
    float* xx_rgb    = xx_pt + BB * PP;          // B*P
    float* g_pt      = xx_rgb + BB * PP;         // B*DX
    float* g_rgb     = g_pt + BB * DX;           // B*DR
    float* nvalid    = g_rgb + BB * DR;          // B
    float* ming_pt   = nvalid + BB;              // B
    float* ming_rgb  = ming_pt + BB;             // B
    float* minD_pt   = ming_rgb + BB;            // B*P
    float* minD_rgb  = minD_pt + BB * PP;        // B*P
    float* term      = minD_rgb + BB * PP;       // B

    // 1) norms (coalesced wave-per-row)
    rownorm_kernel<<<(CC * MM) / WAVES, 256, 0, stream>>>(bpt, norms_pt, CC * MM, DX);
    rownorm_kernel<<<(CC * MM) / WAVES, 256, 0, stream>>>(brg, norms_rgb, CC * MM, DR);
    rownorm_kernel<<<(BB * PP) / WAVES, 256, 0, stream>>>(fxyz, xx_pt, BB * PP, DX);
    rownorm_kernel<<<(BB * PP) / WAVES, 256, 0, stream>>>(frgb, xx_rgb, BB * PP, DR);

    // 2) masked means + nvalid
    meanprep_kernel<<<BB, 256, 0, stream>>>(fxyz, frgb, msk, g_pt, g_rgb, nvalid);

    // 3) main WMMA chamfer-min kernels (7 blocks/sample cover 49 M-tiles)
    {
        const int mtiles = PP / 16;                              // 49
        dim3 grid((mtiles + GWAVES - 1) / GWAVES, BB);           // (7, 16)
        gemm_min_kernel<<<grid, 32 * GWAVES, 0, stream>>>(fxyz, bpt, norms_pt,
                                                          xx_pt, lbl, minD_pt, DX);
        gemm_min_kernel<<<grid, 32 * GWAVES, 0, stream>>>(frgb, brg, norms_rgb,
                                                          xx_rgb, lbl, minD_rgb, DR);
    }

    // 4) global-feature branch
    global_min_kernel<<<BB, 256, 0, stream>>>(g_pt,  bgp, lbl, ming_pt,  DX);
    global_min_kernel<<<BB, 256, 0, stream>>>(g_rgb, bgr, lbl, ming_rgb, DR);

    // 5) combine
    sample_term_kernel<<<BB, 256, 0, stream>>>(minD_pt, minD_rgb, msk, nvalid,
                                               ming_pt, ming_rgb, term);
    final_kernel<<<1, 32, 0, stream>>>(term, out);
}
